// AttnConvolutionalDecoder_60464549593311
// MI455X (gfx1250) — matrix operand order repaired and checked by reading, not compiled, not run on hardware
//
#include <hip/hip_runtime.h>
#include <stdint.h>
#include <math.h>

// Problem constants (match reference)
#define Lc 4
#define Kc 3
#define Cc 512
#define Dc 512
#define Ec 512
#define Tc 1024
#define Bc 16
#define Sc 512
#define Vc 32

typedef __attribute__((ext_vector_type(16))) __bf16 v16bf;
typedef __attribute__((ext_vector_type(8)))  float  v8f;

union Frag { uint32_t u[8]; v16bf v; };

__device__ __forceinline__ uint16_t f2bf(float f) {
  union { float f; uint32_t u; } c; c.f = f;
  uint32_t r = c.u + 0x7FFFu + ((c.u >> 16) & 1u);   // round-to-nearest-even
  return (uint16_t)(r >> 16);
}
__device__ __forceinline__ float bf2f(uint16_t h) {
  union { uint32_t u; float f; } c; c.u = ((uint32_t)h) << 16;
  return c.f;
}

__device__ __forceinline__ v8f wmma_bf16(v16bf a, v16bf b, v8f c) {
  // D = A(16x32 bf16) * B(32x16 bf16) + C(16x16 f32)
  return __builtin_amdgcn_wmma_f32_16x16x32_bf16(false, a, false, b, (short)0, c, false, false);
}

// ---- Fragment builders (CDNA5 wave32 WMMA VGPR layouts, ISA 7.12.2) ----
// All fragments are two 16-byte vector loads: dword pairs are contiguous in
// memory for both the A layout (K pairs 0..7 / 16..23 per lane-half) and the
// B layout (same K pairs 0..7 / 16..23 per lane-half along a weight row).

__device__ __forceinline__ v16bf frag_from2(uint4 x, uint4 y) {
  Frag f;
  f.u[0] = x.x; f.u[1] = x.y; f.u[2] = x.z; f.u[3] = x.w;
  f.u[4] = y.x; f.u[5] = y.y; f.u[6] = y.z; f.u[7] = y.w;
  return f.v;
}

// A fragment from LDS tile (rows of lda bf16), row m, K offset k0.
__device__ __forceinline__ v16bf frag_a_lds(const uint16_t* sT, int lda, int m,
                                            int k0, int hi) {
  const uint16_t* p = sT + (size_t)m * lda + k0 + 8 * hi;
  return frag_from2(*(const uint4*)p, *(const uint4*)(p + 16));
}

// A fragment straight from global row-major tile (Arow0 = first tile row).
__device__ __forceinline__ v16bf frag_a_glb(const uint16_t* Arow0, int lda,
                                            int k0, int lane) {
  int m = lane & 15, hi = lane >> 4;
  const uint16_t* p = Arow0 + (size_t)m * lda + k0 + 8 * hi;
  return frag_from2(*(const uint4*)p, *(const uint4*)(p + 16));
}

// B operand for C = A * W^T, W row-major (N,K). Pass Wn0 = W + n0*ldw.
__device__ __forceinline__ v16bf frag_b_glb(const uint16_t* Wn0, int ldw,
                                            int k0, int lane) {
  int n = lane & 15, hi = lane >> 4;
  const uint16_t* p = Wn0 + (size_t)n * ldw + k0 + 8 * hi;
  return frag_from2(*(const uint4*)p, *(const uint4*)(p + 16));
}

// Stage a contiguous block of bf16 (nchunks x 16B) from global into LDS.
__device__ __forceinline__ void stage(const uint16_t* __restrict__ g,
                                      uint16_t* __restrict__ s,
                                      int nchunks, int tid) {
  for (int i = tid; i < nchunks; i += 256)
    *(uint4*)(s + (size_t)i * 8) = *(const uint4*)(g + (size_t)i * 8);
}

// ======================= setup kernels =======================

__global__ __launch_bounds__(256) void k_cvt(const float* __restrict__ in,
                                             uint16_t* __restrict__ out, int n) {
  int i = blockIdx.x * 256 + threadIdx.x;
  if (i < n) out[i] = f2bf(in[i]);
}

// conv weights (L,Cout,Cin,K) f32 -> (L,K,Cout,Cin) bf16 (per-tap contiguous)
__global__ __launch_bounds__(256) void k_cvt_convw(const float* __restrict__ in,
                                                   uint16_t* __restrict__ out, int n) {
  int i = blockIdx.x * 256 + threadIdx.x;
  if (i >= n) return;
  int ci = i % Cc;
  int o  = (i / Cc) % Cc;
  int k  = (i / (Cc * Cc)) % Kc;
  int l  = i / (Cc * Cc * Kc);
  out[i] = f2bf(in[(((size_t)l * Cc + o) * Cc + ci) * Kc + k]);
}

// emb[b,t,d] = label_embed[labels[t,b]] + time_embed[t]; seeds h0 = emb
__global__ __launch_bounds__(256) void k_embed(const int* __restrict__ labels,
                                               const float* __restrict__ lemb,
                                               const float* __restrict__ temb,
                                               uint16_t* __restrict__ embB,
                                               uint16_t* __restrict__ hB) {
  int i = blockIdx.x * 256 + threadIdx.x;  // over B*T*D
  int d = i % Dc;
  int t = (i / Dc) % Tc;
  int b = i / (Dc * Tc);
  int lab = labels[t * Bc + b];
  uint16_t v = f2bf(lemb[(size_t)lab * Dc + d] + temb[(size_t)t * Dc + d]);
  embB[i] = v;
  hB[i]   = v;
}

// enc_seq (S,B,E) f32 -> encT (B,E,S) bf16  (so Gram = encT @ encT^T is
// fully contiguous for both WMMA operands)
__global__ __launch_bounds__(256) void k_encT(const float* __restrict__ enc,
                                              uint16_t* __restrict__ encT) {
  int i = blockIdx.x * 256 + threadIdx.x;  // over B*E*S, contiguous in s
  int s = i % Sc;
  int e = (i / Sc) % Ec;
  int b = i / (Sc * Ec);
  encT[i] = f2bf(enc[((size_t)s * Bc + b) * Ec + e]);
}

// esum[b,e] = sum_s enc_seq[s,b,e] (f32, exact from f32 input)
__global__ __launch_bounds__(256) void k_esum(const float* __restrict__ enc,
                                              float* __restrict__ esum) {
  int i = blockIdx.x * 256 + threadIdx.x;  // over B*E
  int e = i % Ec;
  int b = i / Ec;
  float s = 0.f;
  for (int sI = 0; sI < Sc; ++sI) s += enc[((size_t)sI * Bc + b) * Ec + e];
  esum[i] = s;
}

// Gram[b] = encT[b] (E x S) @ encT[b]^T. Block: 16 Gram rows x 128 cols.
__global__ __launch_bounds__(256) void k_gram(const uint16_t* __restrict__ encT,
                                              uint16_t* __restrict__ MbB) {
  __shared__ __align__(16) uint16_t sA[16 * Sc];   // 16KB
  int tid = threadIdx.x;
  int wv = tid >> 5, lane = tid & 31;
  int b   = blockIdx.x >> 7;                   // 32 row-tiles * 4 col-groups
  int rem = blockIdx.x & 127;
  int m0  = (rem >> 2) << 4;
  int n0  = ((rem & 3) * 8 + wv) << 4;
  const uint16_t* Eb = encT + (size_t)b * Ec * Sc;
  stage(Eb + (size_t)m0 * Sc, sA, 16 * Sc / 8, tid);
  __syncthreads();
  int m = lane & 15, hi = lane >> 4;
  v8f acc = {};
  for (int k = 0; k < Sc; k += 32) {
    acc = wmma_bf16(frag_a_lds(sA, Sc, m, k, hi),
                    frag_b_glb(Eb + (size_t)n0 * Sc, Sc, k, lane), acc);
  }
  int n = lane & 15;
#pragma unroll
  for (int r = 0; r < 8; ++r) {
    int mm = r + 8 * hi;
    MbB[(size_t)b * Ec * Ec + (size_t)(m0 + mm) * Ec + n0 + n] = f2bf(acc[r]);
  }
}

// ======================= per-layer kernels =======================

// conv_out = glu_x * sigmoid(glu_g) + h @ res^T + biases.
// Block: rows t0..t0+15 of one batch (staged with 2 causal halo rows), 128 cols.
__global__ __launch_bounds__(256) void k_conv(const uint16_t* __restrict__ hB,
                                              const uint16_t* __restrict__ wg,  // (K,C,C)
                                              const uint16_t* __restrict__ wi,  // (K,C,C)
                                              const uint16_t* __restrict__ wr,  // (C,C)
                                              const float* __restrict__ bg,
                                              const float* __restrict__ bi,
                                              const float* __restrict__ br,
                                              float* __restrict__ convF,
                                              uint16_t* __restrict__ convB) {
  __shared__ __align__(16) uint16_t sH[18 * Cc];   // 18KB: rows t0-2..t0+15
  int tid = threadIdx.x;
  int wv = tid >> 5, lane = tid & 31;
  int row_tile = blockIdx.x >> 2;
  int n0 = (((blockIdx.x & 3) * 8) + wv) << 4;
  int b  = row_tile >> 6;
  int t0 = (row_tile & 63) << 4;
  const uint16_t* Hb = hB + (size_t)b * Tc * Cc;
  // Stage 18 rows (zero-fill causal rows before t=0)
  for (int i = tid; i < 18 * (Cc / 8); i += 256) {
    int r = i >> 6, c = (i & 63) * 8;
    int t = t0 - 2 + r;
    uint4 v = {0u, 0u, 0u, 0u};
    if (t >= 0) v = *(const uint4*)(Hb + (size_t)t * Cc + c);
    *(uint4*)(sH + (size_t)r * Cc + c) = v;
  }
  __syncthreads();
  int m = lane & 15, hi = lane >> 4;
  v8f ax = {}, ag = {}, ar = {};
  for (int k = 0; k < Cc; k += 32) {
    v16bf a0 = frag_a_lds(sH, Cc, m + 0, k, hi);  // t-2 (tap 0)
    v16bf a1 = frag_a_lds(sH, Cc, m + 1, k, hi);  // t-1 (tap 1)
    v16bf a2 = frag_a_lds(sH, Cc, m + 2, k, hi);  // t   (tap 2)
    const uint16_t* g0 = wg + (size_t)n0 * Cc;
    const uint16_t* i0 = wi + (size_t)n0 * Cc;
    if (k + 32 < Cc) {  // prefetch next weight K-slice (global_prefetch_b8)
      __builtin_prefetch(g0 + (size_t)(lane & 15) * Cc + k + 32, 0, 1);
      __builtin_prefetch(i0 + (size_t)(lane & 15) * Cc + k + 32, 0, 1);
    }
    ax = wmma_bf16(a0, frag_b_glb(g0 + 0 * Cc * Cc, Cc, k, lane), ax);
    ax = wmma_bf16(a1, frag_b_glb(g0 + 1 * Cc * Cc, Cc, k, lane), ax);
    ax = wmma_bf16(a2, frag_b_glb(g0 + 2 * Cc * Cc, Cc, k, lane), ax);
    ag = wmma_bf16(a0, frag_b_glb(i0 + 0 * Cc * Cc, Cc, k, lane), ag);
    ag = wmma_bf16(a1, frag_b_glb(i0 + 1 * Cc * Cc, Cc, k, lane), ag);
    ag = wmma_bf16(a2, frag_b_glb(i0 + 2 * Cc * Cc, Cc, k, lane), ag);
    ar = wmma_bf16(a2, frag_b_glb(wr + (size_t)n0 * Cc, Cc, k, lane), ar);
  }
  int n = lane & 15;
  float bgx = bg[n0 + n], bgi = bi[n0 + n], bgr = br[n0 + n];
#pragma unroll
  for (int r = 0; r < 8; ++r) {
    int mm = r + 8 * hi;
    size_t row = (size_t)b * Tc + t0 + mm;
    float xv = ax[r] + bgx;
    float gv = ag[r] + bgi;
    float rv = ar[r] + bgr;
    float co = xv * (1.f / (1.f + __expf(-gv))) + rv;
    convF[row * Cc + n0 + n] = co;
    convB[row * Cc + n0 + n] = f2bf(co);
  }
}

// d = conv_out @ in2enc^T + emb @ lab2enc^T + biases
__global__ __launch_bounds__(256) void k_dgemm(const uint16_t* __restrict__ convB,
                                               const uint16_t* __restrict__ embB,
                                               const uint16_t* __restrict__ w1, // (E,C)
                                               const uint16_t* __restrict__ w2, // (E,D)
                                               const float* __restrict__ b1,
                                               const float* __restrict__ b2,
                                               uint16_t* __restrict__ dB) {
  __shared__ __align__(16) uint16_t sA[16 * Cc];
  __shared__ __align__(16) uint16_t sE[16 * Dc];
  int tid = threadIdx.x;
  int wv = tid >> 5, lane = tid & 31;
  int row_tile = blockIdx.x >> 2;
  int n0 = (((blockIdx.x & 3) * 8) + wv) << 4;
  size_t r0 = (size_t)row_tile * 16;
  stage(convB + r0 * Cc, sA, 16 * Cc / 8, tid);
  stage(embB + r0 * Dc, sE, 16 * Dc / 8, tid);
  __syncthreads();
  int m = lane & 15, hi = lane >> 4;
  v8f acc = {};
  for (int k = 0; k < Cc; k += 32) {
    acc = wmma_bf16(frag_a_lds(sA, Cc, m, k, hi),
                    frag_b_glb(w1 + (size_t)n0 * Cc, Cc, k, lane), acc);
    acc = wmma_bf16(frag_a_lds(sE, Dc, m, k, hi),
                    frag_b_glb(w2 + (size_t)n0 * Dc, Dc, k, lane), acc);
  }
  int n = lane & 15;
  float bias = b1[n0 + n] + b2[n0 + n];
#pragma unroll
  for (int r = 0; r < 8; ++r) {
    int mm = r + 8 * hi;
    dB[(r0 + mm) * Ec + n0 + n] = f2bf(acc[r] + bias);
  }
}

// denom[row] = d[row,:] . esum[b,:]  (vectorized bf16 reads)
__global__ __launch_bounds__(256) void k_denom(const uint16_t* __restrict__ dB,
                                               const float* __restrict__ esum,
                                               float* __restrict__ denom) {
  int row = blockIdx.x * 256 + threadIdx.x;  // B*T rows
  int b = row >> 10;
  const uint16_t* dr = dB + (size_t)row * Ec;
  const float* es = esum + (size_t)b * Ec;
  float s = 0.f;
  for (int e = 0; e < Ec; e += 8) {
    uint4 v = *(const uint4*)(dr + e);
    uint32_t w[4] = {v.x, v.y, v.z, v.w};
#pragma unroll
    for (int j = 0; j < 4; ++j) {
      s += bf2f((uint16_t)(w[j] & 0xFFFFu)) * es[e + 2 * j];
      s += bf2f((uint16_t)(w[j] >> 16)) * es[e + 2 * j + 1];
    }
  }
  denom[row] = s;
}

// ctx = (d @ Gram[b]) / denom   (Gram symmetric -> W-style B operand valid)
__global__ __launch_bounds__(256) void k_numctx(const uint16_t* __restrict__ dB,
                                                const uint16_t* __restrict__ MbB,
                                                const float* __restrict__ denom,
                                                uint16_t* __restrict__ ctxB) {
  __shared__ __align__(16) uint16_t sA[16 * Ec];
  int tid = threadIdx.x;
  int wv = tid >> 5, lane = tid & 31;
  int row_tile = blockIdx.x >> 2;
  int n0 = (((blockIdx.x & 3) * 8) + wv) << 4;
  size_t r0 = (size_t)row_tile * 16;
  int b = row_tile >> 6;
  const uint16_t* Mb = MbB + (size_t)b * Ec * Ec;
  stage(dB + r0 * Ec, sA, 16 * Ec / 8, tid);
  __syncthreads();
  int m = lane & 15, hi = lane >> 4;
  v8f acc = {};
  for (int k = 0; k < Ec; k += 32) {
    acc = wmma_bf16(frag_a_lds(sA, Ec, m, k, hi),
                    frag_b_glb(Mb + (size_t)n0 * Ec, Ec, k, lane), acc);
  }
  int n = lane & 15;
#pragma unroll
  for (int r = 0; r < 8; ++r) {
    int mm = r + 8 * hi;
    float inv = 1.f / denom[r0 + mm];
    ctxB[(r0 + mm) * Ec + n0 + n] = f2bf(acc[r] * inv);
  }
}

// h_next = ctx @ enc2in^T + emb @ inres^T + conv_out + biases
__global__ __launch_bounds__(256) void k_hnext(const uint16_t* __restrict__ ctxB,
                                               const uint16_t* __restrict__ embB,
                                               const uint16_t* __restrict__ w1, // (C,E)
                                               const uint16_t* __restrict__ w2, // (C,D)
                                               const float* __restrict__ b1,
                                               const float* __restrict__ b2,
                                               const float* __restrict__ convF,
                                               uint16_t* __restrict__ hB) {
  __shared__ __align__(16) uint16_t sA[16 * Ec];
  __shared__ __align__(16) uint16_t sE[16 * Dc];
  int tid = threadIdx.x;
  int wv = tid >> 5, lane = tid & 31;
  int row_tile = blockIdx.x >> 2;
  int n0 = (((blockIdx.x & 3) * 8) + wv) << 4;
  size_t r0 = (size_t)row_tile * 16;
  stage(ctxB + r0 * Ec, sA, 16 * Ec / 8, tid);
  stage(embB + r0 * Dc, sE, 16 * Dc / 8, tid);
  __syncthreads();
  int m = lane & 15, hi = lane >> 4;
  v8f acc = {};
  for (int k = 0; k < Ec; k += 32) {
    acc = wmma_bf16(frag_a_lds(sA, Ec, m, k, hi),
                    frag_b_glb(w1 + (size_t)n0 * Ec, Ec, k, lane), acc);
    acc = wmma_bf16(frag_a_lds(sE, Dc, m, k, hi),
                    frag_b_glb(w2 + (size_t)n0 * Dc, Dc, k, lane), acc);
  }
  int n = lane & 15;
  float bias = b1[n0 + n] + b2[n0 + n];
#pragma unroll
  for (int r = 0; r < 8; ++r) {
    int mm = r + 8 * hi;
    size_t row = r0 + mm;
    hB[row * Cc + n0 + n] = f2bf(acc[r] + bias + convF[row * Cc + n0 + n]);
  }
}

// out(T,B,V) = h @ out_proj^T + emb @ out_res^T + biases (V=32: 2 col tiles)
__global__ __launch_bounds__(256) void k_out(const uint16_t* __restrict__ hB,
                                             const uint16_t* __restrict__ embB,
                                             const uint16_t* __restrict__ wp, // (V,C)
                                             const uint16_t* __restrict__ wr, // (V,D)
                                             const float* __restrict__ bp,
                                             const float* __restrict__ br,
                                             float* __restrict__ out) {
  int wave = blockIdx.x * 8 + (threadIdx.x >> 5);
  int lane = threadIdx.x & 31;
  int row_tile = wave >> 1;
  int n0 = (wave & 1) << 4;
  size_t r0 = (size_t)row_tile * 16;
  v8f acc = {};
  for (int k = 0; k < Cc; k += 32) {
    acc = wmma_bf16(frag_a_glb(hB + r0 * Cc, Cc, k, lane),
                    frag_b_glb(wp + (size_t)n0 * Cc, Cc, k, lane), acc);
    acc = wmma_bf16(frag_a_glb(embB + r0 * Dc, Dc, k, lane),
                    frag_b_glb(wr + (size_t)n0 * Dc, Dc, k, lane), acc);
  }
  int n = lane & 15, hi = lane >> 4;
  float bias = bp[n0 + n] + br[n0 + n];
#pragma unroll
  for (int r = 0; r < 8; ++r) {
    int m = r + 8 * hi;
    size_t row = r0 + m;               // row = b*T + t
    int b = (int)(row >> 10);
    int t = (int)(row & 1023);
    out[((size_t)t * Bc + b) * Vc + n0 + n] = acc[r] + bias;
  }
}

// ======================= host =======================

extern "C" void kernel_launch(void* const* d_in, const int* in_sizes, int n_in,
                              void* d_out, int out_size, void* d_ws, size_t ws_size,
                              hipStream_t stream) {
  const int*   labels    = (const int*)  d_in[0];
  const float* enc_seq   = (const float*)d_in[1];
  const float* lemb      = (const float*)d_in[2];
  const float* temb      = (const float*)d_in[3];
  const float* conv_glu_w= (const float*)d_in[4];
  const float* conv_glu_b= (const float*)d_in[5];
  const float* conv_id_w = (const float*)d_in[6];
  const float* conv_id_b = (const float*)d_in[7];
  const float* res_w     = (const float*)d_in[8];
  const float* res_b     = (const float*)d_in[9];
  const float* inres_w   = (const float*)d_in[10];
  const float* inres_b   = (const float*)d_in[11];
  const float* in2enc_w  = (const float*)d_in[12];
  const float* in2enc_b  = (const float*)d_in[13];
  const float* lab2enc_w = (const float*)d_in[14];
  const float* lab2enc_b = (const float*)d_in[15];
  const float* enc2in_w  = (const float*)d_in[16];
  const float* enc2in_b  = (const float*)d_in[17];
  const float* outres_w  = (const float*)d_in[18];
  const float* outres_b  = (const float*)d_in[19];
  const float* outproj_w = (const float*)d_in[20];
  const float* outproj_b = (const float*)d_in[21];
  float* out = (float*)d_out;

  char* ws = (char*)d_ws;
  size_t off = 0;
  auto alloc = [&](size_t bytes) -> void* {
    void* p = ws + off;
    off += (bytes + 255) & ~(size_t)255;
    return p;
  };
  const size_t NTOK = (size_t)Bc * Tc;          // 16384 rows
  uint16_t* embB = (uint16_t*)alloc(NTOK * Dc * 2);
  uint16_t* hB   = (uint16_t*)alloc(NTOK * Cc * 2);
  uint16_t* convB= (uint16_t*)alloc(NTOK * Cc * 2);
  uint16_t* dB   = (uint16_t*)alloc(NTOK * Ec * 2);
  uint16_t* ctxB = (uint16_t*)alloc(NTOK * Ec * 2);
  uint16_t* encT = (uint16_t*)alloc((size_t)Bc * Ec * Sc * 2);
  uint16_t* MbB  = (uint16_t*)alloc((size_t)Bc * Ec * Ec * 2);
  uint16_t* wgB  = (uint16_t*)alloc((size_t)Lc * Kc * Cc * Cc * 2);
  uint16_t* wiB  = (uint16_t*)alloc((size_t)Lc * Kc * Cc * Cc * 2);
  uint16_t* wrB  = (uint16_t*)alloc((size_t)Lc * Cc * Cc * 2);
  uint16_t* winresB  = (uint16_t*)alloc((size_t)Lc * Cc * Dc * 2);
  uint16_t* win2encB = (uint16_t*)alloc((size_t)Lc * Ec * Cc * 2);
  uint16_t* wlab2encB= (uint16_t*)alloc((size_t)Lc * Ec * Dc * 2);
  uint16_t* wenc2inB = (uint16_t*)alloc((size_t)Lc * Cc * Ec * 2);
  uint16_t* woutresB = (uint16_t*)alloc((size_t)Vc * Dc * 2);
  uint16_t* woutprojB= (uint16_t*)alloc((size_t)Vc * Cc * 2);
  float* convF = (float*)alloc(NTOK * Cc * 4);
  float* esum  = (float*)alloc((size_t)Bc * Ec * 4);
  float* denom = (float*)alloc(NTOK * 4);
  (void)ws_size; (void)n_in; (void)in_sizes; (void)out_size;

  auto cvt = [&](const float* src, uint16_t* dst, int n) {
    k_cvt<<<(n + 255) / 256, 256, 0, stream>>>(src, dst, n);
  };

  // Weight conversion (f32 -> bf16, conv weights re-laid per tap)
  {
    int n = Lc * Kc * Cc * Cc;
    k_cvt_convw<<<(n + 255) / 256, 256, 0, stream>>>(conv_glu_w, wgB, n);
    k_cvt_convw<<<(n + 255) / 256, 256, 0, stream>>>(conv_id_w, wiB, n);
  }
  cvt(res_w,     wrB,       Lc * Cc * Cc);
  cvt(inres_w,   winresB,   Lc * Cc * Dc);
  cvt(in2enc_w,  win2encB,  Lc * Ec * Cc);
  cvt(lab2enc_w, wlab2encB, Lc * Ec * Dc);
  cvt(enc2in_w,  wenc2inB,  Lc * Cc * Ec);
  cvt(outres_w,  woutresB,  Vc * Dc);
  cvt(outproj_w, woutprojB, Vc * Cc);

  // Embedding (also h0), encoder transpose, esum, per-batch Gram matrices
  k_embed<<<(int)(NTOK * Dc / 256), 256, 0, stream>>>(labels, lemb, temb, embB, hB);
  k_encT<<<(Bc * Ec * Sc) / 256, 256, 0, stream>>>(enc_seq, encT);
  k_esum<<<(Bc * Ec) / 256, 256, 0, stream>>>(enc_seq, esum);
  k_gram<<<Bc * 128, 256, 0, stream>>>(encT, MbB);   // 32 row-tiles * 4 col-groups

  const int GEMM_BLOCKS = (int)((NTOK / 16) * 4);    // row-tiles * 4 col-groups
  for (int l = 0; l < Lc; ++l) {
    k_conv<<<GEMM_BLOCKS, 256, 0, stream>>>(
        hB,
        wgB + (size_t)l * Kc * Cc * Cc,
        wiB + (size_t)l * Kc * Cc * Cc,
        wrB + (size_t)l * Cc * Cc,
        conv_glu_b + (size_t)l * Cc,
        conv_id_b + (size_t)l * Cc,
        res_b + (size_t)l * Cc,
        convF, convB);
    k_dgemm<<<GEMM_BLOCKS, 256, 0, stream>>>(
        convB, embB,
        win2encB + (size_t)l * Ec * Cc,
        wlab2encB + (size_t)l * Ec * Dc,
        in2enc_b + (size_t)l * Ec,
        lab2enc_b + (size_t)l * Ec,
        dB);
    k_denom<<<(int)(NTOK / 256), 256, 0, stream>>>(dB, esum, denom);
    k_numctx<<<GEMM_BLOCKS, 256, 0, stream>>>(dB, MbB, denom, ctxB);
    k_hnext<<<GEMM_BLOCKS, 256, 0, stream>>>(
        ctxB, embB,
        wenc2inB + (size_t)l * Cc * Ec,
        winresB + (size_t)l * Cc * Dc,
        enc2in_b + (size_t)l * Cc,
        inres_b + (size_t)l * Cc,
        convF, hB);
  }
  k_out<<<(int)((NTOK / 16) * (Vc / 16) / 8), 256, 0, stream>>>(
      hB, embB, woutprojB, woutresB, outproj_b, outres_b, out);
}
